// MultiHeadAttention_49709951484276
// MI455X (gfx1250) — matrix operand order repaired and checked by reading, not compiled, not run on hardware
//
#include <hip/hip_runtime.h>

typedef __attribute__((ext_vector_type(16))) _Float16 v16h;
typedef __attribute__((ext_vector_type(8)))  _Float16 v8h;
typedef __attribute__((ext_vector_type(8)))  float    v8f;

#define D_MODEL 1024
#define NH      16
#define DK      64
#define SEQ     2048
#define BATCH   4

#define WMMA_F16(a, b, c) \
  __builtin_amdgcn_wmma_f32_16x16x32_f16(false, (a), false, (b), (short)0, (c), false, false)

// Async global->LDS staging (gfx1250). Probe-confirmed param type:
// 'int __vector(4) __device__ *' = int4 addrspace(1) src / addrspace(3) dst.
#if defined(__gfx1250__) && \
    __has_builtin(__builtin_amdgcn_global_load_async_to_lds_b128) && \
    __has_builtin(__builtin_amdgcn_s_wait_asynccnt)
#define HAVE_ASYNC_LDS 1
typedef __attribute__((vector_size(16))) int v4i_t;
typedef __attribute__((address_space(1))) v4i_t gv4i_t;
typedef __attribute__((address_space(3))) v4i_t lv4i_t;
#else
#define HAVE_ASYNC_LDS 0
#endif

// ---------------------------------------------------------------------------
// WMMA fragment loaders (CDNA5 wave32; contiguous k order, same in A and B)
// ---------------------------------------------------------------------------

// A-matrix 16x32 f16, this code's contiguous k order (not the ISA table): lanes 0-15 hold row M=lane, K in {0..15};
// lanes 16-31 hold row M=lane-16, K in {16..31}. VGPR i packs 2 K's.
static __device__ __forceinline__ v16h load_a_frag(const _Float16* A, int lda) {
  int lane = threadIdx.x & 31;
  int m    = lane & 15;
  int hf   = lane >> 4;
  const _Float16* row = A + m * lda;
  v16h a;
#pragma unroll
  for (int i = 0; i < 8; ++i) {
    int kb = 2 * i + 16 * hf;
    a[2 * i]     = row[kb];
    a[2 * i + 1] = row[kb + 1];
  }
  return a;
}

// B-operand (32x16, K x N) built from a ROW-MAJOR matrix holding B's columns
// as rows (B[k][n] = Tp[n*ldk + k]): per lane one contiguous 32-byte chunk.
static __device__ __forceinline__ v16h load_bt_frag(const _Float16* Tp, int ldk) {
  int lane = threadIdx.x & 31;
  int n    = lane & 15;
  int koff = (lane >> 4) * 16;
  return *(const v16h*)(Tp + n * ldk + koff);
}

// ---------------------------------------------------------------------------
// f32 -> f16 conversion (activations)
// ---------------------------------------------------------------------------
__global__ void cvt_f32_f16(const float* __restrict__ s, _Float16* __restrict__ d, int n) {
  int i      = blockIdx.x * blockDim.x + threadIdx.x;
  int stride = gridDim.x * blockDim.x;
  for (; i < n; i += stride) d[i] = (_Float16)s[i];
}

// f32 [K,N] -> f16 transposed [N,K] (weights), 32x32 tiles through LDS
__global__ void __launch_bounds__(256) cvt_transpose(const float* __restrict__ src,
                                                     _Float16* __restrict__ dst,
                                                     int K, int N) {
  __shared__ float tile[32][33];
  int nbx = N >> 5;
  int bx  = blockIdx.x % nbx;   // column tile
  int by  = blockIdx.x / nbx;   // row tile
  int tx  = threadIdx.x & 31;
  int ty  = threadIdx.x >> 5;   // 0..7
#pragma unroll
  for (int i = ty; i < 32; i += 8)
    tile[i][tx] = src[(by * 32 + i) * N + bx * 32 + tx];
  __syncthreads();
#pragma unroll
  for (int i = ty; i < 32; i += 8)
    dst[(bx * 32 + i) * K + by * 32 + tx] = (_Float16)tile[tx][i];
}

// ---------------------------------------------------------------------------
// Blocked GEMM: out[M,N] = A[M,K](f16) @ Bt[N,K]^T(f16) + bias.
// Block = 256 threads (8 waves), tile 128x64; each wave computes 32x32
// (2x2 WMMA register block). Double-buffered LDS staging: async loads for
// stage i+1 are issued before waiting on stage i (s_wait_asynccnt 3 keeps
// the next stage's 3 loads in flight while compute runs).
// MODE: 0 = f32 row-major, 1 = f16 row-major, 2 = f16 head-transposed
//       (dst[((row/SEQ)*1024 + col)*SEQ + row%SEQ], i.e. [B,H,DK,SEQ]).
// ---------------------------------------------------------------------------
template <int MODE>
__global__ void __launch_bounds__(256) gemm_wmma(const _Float16* __restrict__ A,
                                                 const _Float16* __restrict__ Bt,
                                                 const float* __restrict__ bias,
                                                 void* __restrict__ outv,
                                                 int M, int N, int K) {
  __shared__ __align__(64) _Float16 As[2][128][32];
  __shared__ __align__(64) _Float16 Bs[2][64][32];

  int t   = threadIdx.x;
  int ntb = N >> 6;                // 64-wide column tiles
  int tm  = blockIdx.x / ntb;
  int tn  = blockIdx.x % ntb;

  // per-thread staging addresses: A = 2 x b128, B = 1 x b128 per stage
  const _Float16* ga = A  + (tm * 128 + (t >> 1)) * K + (t & 1) * 16;
  const _Float16* gb = Bt + (tn * 64 + (t >> 2)) * K + (t & 3) * 8;
  int la_row = t >> 1, la_col = (t & 1) * 16;
  int lb_row = t >> 2, lb_col = (t & 3) * 8;

  auto stage = [&](int buf, int k0) {
    const _Float16* a  = ga + k0;
    const _Float16* b  = gb + k0;
    _Float16*       pa = &As[buf][la_row][la_col];
    _Float16*       pb = &Bs[buf][lb_row][lb_col];
#if HAVE_ASYNC_LDS
    __builtin_amdgcn_global_load_async_to_lds_b128((gv4i_t*)a,       (lv4i_t*)pa,       0, 0);
    __builtin_amdgcn_global_load_async_to_lds_b128((gv4i_t*)(a + 8), (lv4i_t*)(pa + 8), 0, 0);
    __builtin_amdgcn_global_load_async_to_lds_b128((gv4i_t*)b,       (lv4i_t*)pb,       0, 0);
#else
    *(v16h*)pa = *(const v16h*)a;
    *(v8h*)pb  = *(const v8h*)b;
#endif
  };

  int w = t >> 5, lane = t & 31;
  int wm = w >> 1, wn = w & 1;     // 4x2 wave grid over the 128x64 tile

  v8f acc00 = {}, acc01 = {}, acc10 = {}, acc11 = {};

  stage(0, 0);
  for (int k0 = 0, it = 0; k0 < K; k0 += 32, ++it) {
    int cur = it & 1;
    if (k0 + 32 < K) {
      stage(cur ^ 1, k0 + 32);
#if HAVE_ASYNC_LDS
      __builtin_amdgcn_s_wait_asynccnt(3);   // current stage done, next in flight
#endif
    } else {
#if HAVE_ASYNC_LDS
      __builtin_amdgcn_s_wait_asynccnt(0);
#endif
    }
    __syncthreads();

    v16h a0 = load_a_frag(&As[cur][32 * wm][0], 32);
    v16h a1 = load_a_frag(&As[cur][32 * wm + 16][0], 32);
    v16h b0 = load_bt_frag(&Bs[cur][32 * wn][0], 32);
    v16h b1 = load_bt_frag(&Bs[cur][32 * wn + 16][0], 32);

    acc00 = WMMA_F16(a0, b0, acc00);
    acc01 = WMMA_F16(a0, b1, acc01);
    acc10 = WMMA_F16(a1, b0, acc10);
    acc11 = WMMA_F16(a1, b1, acc11);
    __syncthreads();   // all reads of buffer `cur` done before it is re-staged
  }

  int   n       = lane & 15;
  int   mb      = (lane >> 4) * 8;
  int   rowBase = tm * 128 + 32 * wm + mb;
  int   colBase = tn * 64 + 32 * wn + n;
  float bv0     = bias[colBase];
  float bv1     = bias[colBase + 16];

  auto oidx = [&](int row, int col) -> int {
    if constexpr (MODE == 2)
      return (((row >> 11) << 10) + col) * SEQ + (row & (SEQ - 1));
    else
      return row * N + col;
  };

#pragma unroll
  for (int r = 0; r < 8; ++r) {
    float v00 = acc00[r] + bv0;
    float v01 = acc01[r] + bv1;
    float v10 = acc10[r] + bv0;
    float v11 = acc11[r] + bv1;
    int   r0  = rowBase + r;
    int   r1  = rowBase + 16 + r;
    if constexpr (MODE == 0) {
      float* o = (float*)outv;
      o[oidx(r0, colBase)]      = v00;
      o[oidx(r0, colBase + 16)] = v01;
      o[oidx(r1, colBase)]      = v10;
      o[oidx(r1, colBase + 16)] = v11;
    } else {
      _Float16* o = (_Float16*)outv;
      o[oidx(r0, colBase)]      = (_Float16)v00;
      o[oidx(r0, colBase + 16)] = (_Float16)v01;
      o[oidx(r1, colBase)]      = (_Float16)v10;
      o[oidx(r1, colBase + 16)] = (_Float16)v11;
    }
  }
}

// ---------------------------------------------------------------------------
// Causal flash attention: one wave handles 16 query rows of one (b,h).
// Keys consumed 32 at a time; P re-laid-out as A-fragment via LDS.
// K is [B,S,D] (row-major per head => Q.K^T fragments contiguous);
// V is head-transposed [B,H,DK,SEQ] => P.V fragments contiguous.
// ---------------------------------------------------------------------------
__global__ void __launch_bounds__(128) attn_wmma(const _Float16* __restrict__ Q,
                                                 const _Float16* __restrict__ Km,
                                                 const _Float16* __restrict__ Vt,
                                                 _Float16* __restrict__ O) {
  __shared__ __align__(64) _Float16 Plds[4][16][32];

  int widx = threadIdx.x >> 5;
  int lane = threadIdx.x & 31;
  int gw   = blockIdx.x * 4 + widx;
  int qblk = gw & 127;          // S/16 = 128 query tiles
  int h    = (gw >> 7) & 15;
  int b    = gw >> 11;

  const _Float16* Qp = Q + (b * SEQ + qblk * 16) * D_MODEL + h * DK;
  v16h a0 = load_a_frag(Qp, D_MODEL);       // features 0..31
  v16h a1 = load_a_frag(Qp + 32, D_MODEL);  // features 32..63

  float mrun[8], lrun[8];
  v8f   o0 = {}, o1 = {}, o2 = {}, o3 = {};
#pragma unroll
  for (int r = 0; r < 8; ++r) { mrun[r] = -1e30f; lrun[r] = 0.0f; }

  int n        = lane & 15;
  int halfId   = lane >> 4;
  int rowgBase = qblk * 16 + halfId * 8;

  const _Float16* Kh  = Km + b * SEQ * D_MODEL + h * DK;
  const _Float16* Vth = Vt + (b * 1024 + h * DK) * SEQ;   // [DK rows][SEQ keys]

  int nj = (qblk >> 1) + 1;  // causal: key blocks of 32 up to diagonal
  for (int j = 0; j < nj; ++j) {
    int kb0 = j * 32;

    if (j + 1 < nj) {  // prefetch next key/value block
      __builtin_prefetch(Kh + (kb0 + 32 + lane) * D_MODEL, 0, 0);
      __builtin_prefetch(Vth + (lane << 1) * SEQ + kb0 + 32, 0, 0);
    }

    // scores: 16x32 tile = two 16x16 WMMA accumulators, K-dim = 64 features
    v16h kA0 = load_bt_frag(Kh + kb0 * D_MODEL,             D_MODEL);
    v16h kA1 = load_bt_frag(Kh + kb0 * D_MODEL + 32,        D_MODEL);
    v16h kB0 = load_bt_frag(Kh + (kb0 + 16) * D_MODEL,      D_MODEL);
    v16h kB1 = load_bt_frag(Kh + (kb0 + 16) * D_MODEL + 32, D_MODEL);

    v8f z  = {};
    v8f sA = WMMA_F16(a0, kA0, z);
    sA     = WMMA_F16(a1, kA1, sA);
    v8f sB = WMMA_F16(a0, kB0, z);
    sB     = WMMA_F16(a1, kB1, sB);

    int   colA = kb0 + n, colB = colA + 16;
    float pA[8], pB[8];
#pragma unroll
    for (int r = 0; r < 8; ++r) {
      int   rowg = rowgBase + r;
      float vA   = (colA <= rowg) ? sA[r] * 0.125f : -1e30f;  // 1/sqrt(64)
      float vB   = (colB <= rowg) ? sB[r] * 0.125f : -1e30f;
      float mx   = fmaxf(vA, vB);
      mx = fmaxf(mx, __shfl_xor(mx, 1, 32));
      mx = fmaxf(mx, __shfl_xor(mx, 2, 32));
      mx = fmaxf(mx, __shfl_xor(mx, 4, 32));
      mx = fmaxf(mx, __shfl_xor(mx, 8, 32));
      float mnew = fmaxf(mrun[r], mx);
      float corr = __expf(mrun[r] - mnew);
      float eA   = __expf(vA - mnew);
      float eB   = __expf(vB - mnew);
      float rs   = eA + eB;
      rs += __shfl_xor(rs, 1, 32);
      rs += __shfl_xor(rs, 2, 32);
      rs += __shfl_xor(rs, 4, 32);
      rs += __shfl_xor(rs, 8, 32);
      lrun[r] = lrun[r] * corr + rs;
      mrun[r] = mnew;
      o0[r] *= corr; o1[r] *= corr; o2[r] *= corr; o3[r] *= corr;
      pA[r] = eA; pB[r] = eB;
    }

    // P (C-layout) -> LDS row-major 16x32 -> reload in A-fragment layout
#pragma unroll
    for (int r = 0; r < 8; ++r) {
      int m = halfId * 8 + r;
      Plds[widx][m][n]      = (_Float16)pA[r];
      Plds[widx][m][16 + n] = (_Float16)pB[r];
    }
    v16h pf = load_a_frag(&Plds[widx][0][0], 32);

    // P(16x32) @ V(32x64): 4 d-tiles, contiguous fragments from Vt
    v16h vf0 = load_bt_frag(Vth + 0  * SEQ + kb0, SEQ);
    v16h vf1 = load_bt_frag(Vth + 16 * SEQ + kb0, SEQ);
    v16h vf2 = load_bt_frag(Vth + 32 * SEQ + kb0, SEQ);
    v16h vf3 = load_bt_frag(Vth + 48 * SEQ + kb0, SEQ);
    o0 = WMMA_F16(pf, vf0, o0);
    o1 = WMMA_F16(pf, vf1, o1);
    o2 = WMMA_F16(pf, vf2, o2);
    o3 = WMMA_F16(pf, vf3, o3);
  }

  _Float16* Op = O + (b * SEQ + qblk * 16) * D_MODEL + h * DK;
#pragma unroll
  for (int r = 0; r < 8; ++r) {
    int   m   = halfId * 8 + r;
    float inv = 1.0f / lrun[r];
    Op[m * D_MODEL + 0  + n] = (_Float16)(o0[r] * inv);
    Op[m * D_MODEL + 16 + n] = (_Float16)(o1[r] * inv);
    Op[m * D_MODEL + 32 + n] = (_Float16)(o2[r] * inv);
    Op[m * D_MODEL + 48 + n] = (_Float16)(o3[r] * inv);
  }
}

// ---------------------------------------------------------------------------
// Launcher
// ---------------------------------------------------------------------------
extern "C" void kernel_launch(void* const* d_in, const int* in_sizes, int n_in,
                              void* d_out, int out_size, void* d_ws, size_t ws_size,
                              hipStream_t stream) {
  const float* q  = (const float*)d_in[0];
  const float* v  = (const float*)d_in[1];
  const float* k  = (const float*)d_in[2];
  const float* Wq = (const float*)d_in[3];
  const float* bq = (const float*)d_in[4];
  const float* Wk = (const float*)d_in[5];
  const float* bk = (const float*)d_in[6];
  const float* Wv = (const float*)d_in[7];
  const float* bv = (const float*)d_in[8];
  const float* Wo = (const float*)d_in[9];
  const float* bo = (const float*)d_in[10];
  float*       out = (float*)d_out;

  const size_t ACT = (size_t)BATCH * SEQ * D_MODEL;  // 8,388,608
  const size_t WEL = (size_t)D_MODEL * D_MODEL;      // 1,048,576

  char* ws = (char*)d_ws;
  _Float16* Xq   = (_Float16*)ws; ws += ACT * 2;
  _Float16* Xk   = (_Float16*)ws; ws += ACT * 2;
  _Float16* Xv   = (_Float16*)ws; ws += ACT * 2;
  _Float16* Wt_q = (_Float16*)ws; ws += WEL * 2;   // transposed [N,K]
  _Float16* Wt_k = (_Float16*)ws; ws += WEL * 2;
  _Float16* Wt_v = (_Float16*)ws; ws += WEL * 2;
  _Float16* Wt_o = (_Float16*)ws; ws += WEL * 2;
  _Float16* Qm   = (_Float16*)ws; ws += ACT * 2;
  _Float16* Km   = (_Float16*)ws; ws += ACT * 2;
  _Float16* Vt   = (_Float16*)ws; ws += ACT * 2;   // [B,H,DK,SEQ]
  _Float16* AO   = Xq;  // reuse: Xq dead after Q projection

  // activations f32 -> f16
  cvt_f32_f16<<<4096, 256, 0, stream>>>(q, Xq, (int)ACT);
  cvt_f32_f16<<<4096, 256, 0, stream>>>(k, Xk, (int)ACT);
  cvt_f32_f16<<<4096, 256, 0, stream>>>(v, Xv, (int)ACT);

  // weights f32 [K,N] -> f16 transposed [N,K]
  const int tBlocks = (D_MODEL / 32) * (D_MODEL / 32);  // 1024
  cvt_transpose<<<tBlocks, 256, 0, stream>>>(Wq, Wt_q, D_MODEL, D_MODEL);
  cvt_transpose<<<tBlocks, 256, 0, stream>>>(Wk, Wt_k, D_MODEL, D_MODEL);
  cvt_transpose<<<tBlocks, 256, 0, stream>>>(Wv, Wt_v, D_MODEL, D_MODEL);
  cvt_transpose<<<tBlocks, 256, 0, stream>>>(Wo, Wt_o, D_MODEL, D_MODEL);

  const int M = BATCH * SEQ;                              // 8192
  const int gemmBlocks = (M / 128) * (D_MODEL / 64);      // 1024

  // projections: Q,K row-major f16; V head-transposed f16
  gemm_wmma<1><<<gemmBlocks, 256, 0, stream>>>(Xq, Wt_q, bq, Qm, M, D_MODEL, D_MODEL);
  gemm_wmma<1><<<gemmBlocks, 256, 0, stream>>>(Xk, Wt_k, bk, Km, M, D_MODEL, D_MODEL);
  gemm_wmma<2><<<gemmBlocks, 256, 0, stream>>>(Xv, Wt_v, bv, Vt, M, D_MODEL, D_MODEL);

  // causal flash attention: B*H*(S/16) = 8192 waves / 4 per block
  attn_wmma<<<2048, 128, 0, stream>>>(Qm, Km, Vt, AO);

  // output projection (f32 out)
  gemm_wmma<0><<<gemmBlocks, 256, 0, stream>>>(AO, Wt_o, bo, (void*)out, M, D_MODEL, D_MODEL);
}
